// MultiHeadDeformableAttention2D_69947837382911
// MI455X (gfx1250) — matrix operand order repaired and checked by reading, not compiled, not run on hardware
//
#include <hip/hip_runtime.h>
#include <hip/hip_bf16.h>
#include <math.h>

// ---------------------------------------------------------------------------
// MultiHeadDeformableAttention2D for MI455X (gfx1250, wave32, WMMA bf16)
// N=8, H=W=64 (L=4096), E=512, HEADS=16, HD=32, P=4
//
// Pipeline:
//  1) query fp32 -> bf16
//  2) pack/transpose weights to bf16 (Wv|Woff|Wa concat, Wo)
//  3) fused GEMM1 [32768x512]x[512x704] via v_wmma_f32_16x16x32_bf16:
//       cols 0-511  -> value  (stored bf16, row-major [n][l][E])
//       cols 512-639-> raw offsets [nh][l][4][2] fp32
//       cols 640-703-> raw attn logits [nh][l][4] fp32
//  4) softmax(P=4) + bilinear sampling (wave per (n,head,l), lane=channel,
//     coalesced 64B bf16 corner gathers) -> mid bf16 (reuses q buffer)
//  5) GEMM3 [32768x512]x[512x512] -> fp32 output
// ---------------------------------------------------------------------------

typedef __attribute__((ext_vector_type(16))) __bf16 v16bf;
typedef __attribute__((ext_vector_type(8)))  float  v8f;
typedef __attribute__((ext_vector_type(4)))  unsigned int u32x4;

union FragU { u32x4 u[2]; v16bf v; };

#define NB    8
#define LTOT  4096          // H*W
#define EE    512
#define NHEAD 16
#define HD    32
#define PP    4
#define NCAT  704           // 512 (value) + 128 (off) + 64 (attn logits)
#define MROWS (NB * LTOT)   // 32768

// ---------------------------------------------------------------- prep: query -> bf16
__global__ void convert_q_kernel(const float4* __restrict__ q, __bf16* __restrict__ qb) {
    size_t i = (size_t)blockIdx.x * blockDim.x + threadIdx.x;   // over 4,194,304 float4s
    float4 v = q[i];
    size_t o = i * 4;
    qb[o + 0] = (__bf16)v.x;
    qb[o + 1] = (__bf16)v.y;
    qb[o + 2] = (__bf16)v.z;
    qb[o + 3] = (__bf16)v.w;
}

// ------------------------------------------------- prep: transpose+concat weights to bf16
__global__ void pack_weights_kernel(const float* __restrict__ Wv,  const float* __restrict__ bv,
                                    const float* __restrict__ Woff,const float* __restrict__ boff,
                                    const float* __restrict__ Wa,  const float* __restrict__ ba,
                                    const float* __restrict__ Wo,
                                    __bf16* __restrict__ Wt,       // [704][512]  B^T for GEMM1
                                    __bf16* __restrict__ Wot,      // [512][512]  Wo^T for GEMM3
                                    float*  __restrict__ bias_cat) // [704]
{
    int tid = blockIdx.x * blockDim.x + threadIdx.x;
    if (tid >= NCAT * EE) return;
    int nc = tid >> 9;          // output column
    int k  = tid & 511;         // K index
    float v;
    if (nc < 512)      v = Wv  [k * 512 + nc];
    else if (nc < 640) v = Woff[k * 128 + (nc - 512)];
    else               v = Wa  [k * 64  + (nc - 640)];
    Wt[tid] = (__bf16)v;
    if (nc < 512) Wot[tid] = (__bf16)Wo[k * 512 + nc];
    if (tid < NCAT)
        bias_cat[tid] = (tid < 512) ? bv[tid] : (tid < 640 ? boff[tid - 512] : ba[tid - 640]);
}

// ---------------------------------------------------------------- WMMA bf16 GEMM
// A: [32768][512] bf16 row-major.  Bt: [Ncols][512] bf16 (transposed).  K = 512.
// One wave computes a 16(M) x 64(N) tile: 4 accumulators share each A fragment.
// mode 0: fused epilogue splitting cols into value(bf16) / offsets / attn-logits
// mode 1: plain row-major fp32 store (+bias)
__global__ void gemm_wmma_bf16(const __bf16* __restrict__ A,
                               const __bf16* __restrict__ Bt,
                               const float*  __restrict__ bias,
                               int mode,
                               __bf16* __restrict__ out_value,  // [32768][512] bf16
                               float* __restrict__ out_off,     // [nh][l][4][2]
                               float* __restrict__ out_attn,    // [nh][l][4]
                               float* __restrict__ out_plain)   // [32768][512] f32
{
    const int wave = threadIdx.x >> 5;
    const int lane = threadIdx.x & 31;
    const int r    = lane & 15;        // A row / B col / D col within tile
    const int hi   = lane >> 4;        // lane group
    const int kbA  = hi * 16;
    const int kbB  = hi * 16;

    const int m0 = (blockIdx.x * 8 + wave) * 16;   // 16-row M tile
    const int n0 = blockIdx.y * 64;                // 64-col N tile

    v8f acc[4] = {};
    const __bf16* arow = A + (size_t)(m0 + r) * EE;

    for (int k0 = 0; k0 < EE; k0 += 32) {
        FragU a;
        a.u[0] = *(const u32x4*)(arow + k0 + kbA);        // K = kbA .. kbA+7
        a.u[1] = *(const u32x4*)(arow + k0 + kbA + 8);    // K = kbA+8 .. kbA+15
#pragma unroll
        for (int s = 0; s < 4; ++s) {
            const __bf16* bp = Bt + (size_t)(n0 + s * 16 + r) * EE + k0 + kbB;
            FragU b;
            b.u[0] = ((const u32x4*)bp)[0];               // K = kbB .. kbB+7
            b.u[1] = ((const u32x4*)bp)[1];               // K = kbB+8 .. kbB+15
            acc[s] = __builtin_amdgcn_wmma_f32_16x16x32_bf16(
                         false, a.v, false, b.v, (short)0, acc[s], false, false);
        }
    }

    // D layout: elem v -> row = hi*8 + v, col = r (within tile)
    const int n     = m0 >> 12;               // image index (tiles never straddle images)
    const int lbase = (m0 & 4095) + hi * 8;   // first of 8 consecutive l values

#pragma unroll
    for (int s = 0; s < 4; ++s) {
        const int nc = n0 + s * 16 + r;
        const float bi = bias[nc];
        if (mode == 1) {
            float* dst = out_plain + (size_t)(m0 + hi * 8) * EE + nc;
#pragma unroll
            for (int v = 0; v < 8; ++v) dst[(size_t)v * EE] = acc[s][v] + bi;
        } else if (nc < 512) {
            // value: row-major [n][l][E], bf16 (halves sampling gather traffic)
            __bf16* dst = out_value + (size_t)(m0 + hi * 8) * EE + nc;
#pragma unroll
            for (int v = 0; v < 8; ++v) dst[(size_t)v * EE] = (__bf16)(acc[s][v] + bi);
        } else if (nc < 640) {
            const int o = nc - 512;
            const int head = o >> 3, p = (o >> 1) & 3, xy = o & 1;
            const size_t nh = (size_t)n * NHEAD + head;
#pragma unroll
            for (int v = 0; v < 8; ++v)
                out_off[((nh * LTOT + lbase + v) * PP + p) * 2 + xy] = acc[s][v] + bi;
        } else {
            const int o = nc - 640;
            const int head = o >> 2, p = o & 3;
            const size_t nh = (size_t)n * NHEAD + head;
#pragma unroll
            for (int v = 0; v < 8; ++v)
                out_attn[(nh * LTOT + lbase + v) * PP + p] = acc[s][v] + bi;
        }
    }
}

// ---------------------------------------------------------------- sampling + softmax
// One wave per (n, head, l); lane = channel (HD == 32 == wave32). Corner gathers of the
// row-major bf16 value matrix are coalesced 64B transactions.
__global__ void sample_kernel(const __bf16* __restrict__ value, // [n][l][512] bf16
                              const float* __restrict__ offw,   // [nh][l][4][2]
                              const float* __restrict__ attnl,  // [nh][l][4]
                              const float* __restrict__ refp,   // [n][l][2]
                              __bf16* __restrict__ mid)         // [n][l][512] bf16
{
    const int wg   = (int)((blockIdx.x * blockDim.x + threadIdx.x) >> 5); // 0..524287
    const int lane = threadIdx.x & 31;
    const int l    = wg & (LTOT - 1);
    const int nh   = wg >> 12;
    const int n    = nh >> 4;
    const int head = nh & 15;

    const float* op = offw  + ((size_t)nh * LTOT + l) * (PP * 2);
    const float* ap = attnl + ((size_t)nh * LTOT + l) * PP;

    // softmax over P=4 (redundant across lanes; loads broadcast from L1)
    const float a0 = ap[0], a1 = ap[1], a2 = ap[2], a3 = ap[3];
    const float mx = fmaxf(fmaxf(a0, a1), fmaxf(a2, a3));
    const float e0 = __expf(a0 - mx), e1 = __expf(a1 - mx),
                e2 = __expf(a2 - mx), e3 = __expf(a3 - mx);
    const float inv = 1.0f / (e0 + e1 + e2 + e3);
    float w[PP] = { e0 * inv, e1 * inv, e2 * inv, e3 * inv };

    const float rx = refp[((size_t)n * LTOT + l) * 2 + 0];
    const float ry = refp[((size_t)n * LTOT + l) * 2 + 1];

    const __bf16* vbase = value + (size_t)n * LTOT * EE + head * HD + lane;
    float acc = 0.0f;

#pragma unroll
    for (int p = 0; p < PP; ++p) {
        const float lx = rx + op[p * 2 + 0];
        const float ly = ry + op[p * 2 + 1];
        // grid = 2*loc-1 ; x = ((gx+1)*64 - 1)*0.5 = 64*loc - 0.5
        const float x = 64.0f * lx - 0.5f;
        const float y = 64.0f * ly - 0.5f;
        const float x0f = floorf(x), y0f = floorf(y);
        const float fx = x - x0f, fy = y - y0f;
        const int x0 = (int)x0f, y0 = (int)y0f;
        const float aw = w[p];
        const float w00 = aw * (1.0f - fx) * (1.0f - fy);
        const float w10 = aw * fx * (1.0f - fy);
        const float w01 = aw * (1.0f - fx) * fy;
        const float w11 = aw * fx * fy;

        if (x0     >= 0 && x0     < 64 && y0     >= 0 && y0     < 64)
            acc += w00 * (float)vbase[(size_t)(y0 * 64 + x0) * EE];
        if (x0 + 1 >= 0 && x0 + 1 < 64 && y0     >= 0 && y0     < 64)
            acc += w10 * (float)vbase[(size_t)(y0 * 64 + x0 + 1) * EE];
        if (x0     >= 0 && x0     < 64 && y0 + 1 >= 0 && y0 + 1 < 64)
            acc += w01 * (float)vbase[(size_t)((y0 + 1) * 64 + x0) * EE];
        if (x0 + 1 >= 0 && x0 + 1 < 64 && y0 + 1 >= 0 && y0 + 1 < 64)
            acc += w11 * (float)vbase[(size_t)((y0 + 1) * 64 + x0 + 1) * EE];
    }

    mid[((size_t)n * LTOT + l) * EE + head * HD + lane] = (__bf16)acc;
}

// ---------------------------------------------------------------- launch
extern "C" void kernel_launch(void* const* d_in, const int* in_sizes, int n_in,
                              void* d_out, int out_size, void* d_ws, size_t ws_size,
                              hipStream_t stream) {
    (void)in_sizes; (void)n_in; (void)out_size; (void)ws_size;

    const float* query = (const float*)d_in[0];
    const float* refp  = (const float*)d_in[1];
    const float* Wv    = (const float*)d_in[2];
    const float* bv    = (const float*)d_in[3];
    const float* Woff  = (const float*)d_in[4];
    const float* boff  = (const float*)d_in[5];
    const float* Wa    = (const float*)d_in[6];
    const float* ba    = (const float*)d_in[7];
    const float* Wo    = (const float*)d_in[8];
    const float* bo    = (const float*)d_in[9];

    char* ws = (char*)d_ws;
    // workspace layout (bytes)  — total ~93.5 MB
    const size_t OFF_QMID = 0;                         // 32768*512 bf16 = 33,554,432 (q, then mid)
    const size_t OFF_VAL  = OFF_QMID + 33554432;       // 32768*512 bf16 = 33,554,432
    const size_t OFF_OFF  = OFF_VAL  + 33554432;       // 128*4096*8 f32 = 16,777,216
    const size_t OFF_ATT  = OFF_OFF  + 16777216;       // 128*4096*4 f32 =  8,388,608
    const size_t OFF_WT   = OFF_ATT  + 8388608;        // 704*512 bf16   =    720,896
    const size_t OFF_WOT  = OFF_WT   + 720896;         // 512*512 bf16   =    524,288
    const size_t OFF_BIAS = OFF_WOT  + 524288;         // 704 f32

    __bf16* qmid   = (__bf16*)(ws + OFF_QMID);
    __bf16* valw   = (__bf16*)(ws + OFF_VAL);
    float*  offw   = (float*) (ws + OFF_OFF);
    float*  attw   = (float*) (ws + OFF_ATT);
    __bf16* Wt     = (__bf16*)(ws + OFF_WT);
    __bf16* Wot    = (__bf16*)(ws + OFF_WOT);
    float*  biascat= (float*) (ws + OFF_BIAS);

    // 1) query fp32 -> bf16 (16,777,216 elems as 4,194,304 float4)
    convert_q_kernel<<<16384, 256, 0, stream>>>((const float4*)query, qmid);

    // 2) pack/transpose weights (704*512 = 360,448 elems)
    pack_weights_kernel<<<(NCAT * EE + 255) / 256, 256, 0, stream>>>(
        Wv, bv, Woff, boff, Wa, ba, Wo, Wt, Wot, biascat);

    // 3) fused projection GEMM: [32768 x 512] x [512 x 704]
    //    grid: 256 M-blocks (8 waves x 16 rows) x 11 N-blocks (64 cols)
    gemm_wmma_bf16<<<dim3(256, 11), 256, 0, stream>>>(
        qmid, Wt, biascat, /*mode=*/0, valw, offw, attw, nullptr);

    // 4) softmax + bilinear sampling + P-reduction -> mid (bf16, reuses q buffer)
    //    524,288 waves = 65,536 blocks x 8 waves
    sample_kernel<<<65536, 256, 0, stream>>>(valw, offw, attw, refp, qmid);

    // 5) output projection GEMM: [32768 x 512] x [512 x 512] -> d_out fp32
    gemm_wmma_bf16<<<dim3(256, 8), 256, 0, stream>>>(
        qmid, Wot, bo, /*mode=*/1, nullptr, nullptr, nullptr, (float*)d_out);
}
